// HaloAttention_28467043238110
// MI455X (gfx1250) — matrix operand order repaired and checked by reading, not compiled, not run on hardware
//
#include <hip/hip_runtime.h>

typedef __bf16 bf16_t;
typedef __attribute__((ext_vector_type(8)))  __bf16 v8bf;
typedef __attribute__((ext_vector_type(16))) __bf16 v16bf;
typedef __attribute__((ext_vector_type(8)))  float  v8f;
typedef int v4i __attribute__((vector_size(16)));   // matches async-LDS builtin param

#if __has_builtin(__builtin_amdgcn_global_load_async_to_lds_b128)
#define HAVE_ASYNC_COPY 1
#else
#define HAVE_ASYNC_COPY 0
#endif

// 16-byte global -> LDS copy. Async (ASYNCcnt-tracked DMA) when available.
static __device__ __forceinline__ void cp_async16(bf16_t* lds_dst, const bf16_t* gsrc) {
#if HAVE_ASYNC_COPY
    __builtin_amdgcn_global_load_async_to_lds_b128(
        (__attribute__((address_space(1))) v4i*)(void*)gsrc,
        (__attribute__((address_space(3))) v4i*)(void*)lds_dst,
        0, 0);
#else
    *(v8bf*)lds_dst = *(const v8bf*)gsrc;
#endif
}

template <int N>
static __device__ __forceinline__ void async_wait() {
#if HAVE_ASYNC_COPY
#if __has_builtin(__builtin_amdgcn_s_wait_asynccnt)
    __builtin_amdgcn_s_wait_asynccnt(N);
#else
    asm volatile("s_wait_asynccnt %0" ::"n"(N) : "memory");
#endif
#endif
}

static __device__ __forceinline__ bf16_t f2bf(float f) {
    unsigned u = __builtin_bit_cast(unsigned, f);
    unsigned r = u + 0x7FFFu + ((u >> 16) & 1u);          // round-to-nearest-even
    unsigned short s = (unsigned short)(r >> 16);
    return __builtin_bit_cast(bf16_t, s);
}

static __device__ __forceinline__ v16bf combine16(v8bf lo, v8bf hi) {
    return __builtin_shufflevector(lo, hi, 0,1,2,3,4,5,6,7,8,9,10,11,12,13,14,15);
}

// A/B fragment load per ISA 7.12.2 16-bit layout (unswizzled, padded-stride LDS):
// lanes 0-15: K = {0..7},{16..23}; lanes 16-31: K = {8..15},{24..31}
static __device__ __forceinline__ v16bf load_frag(const bf16_t* rowp, int klocal) {
    v8bf lo = *(const v8bf*)(rowp + klocal);
    v8bf hi = *(const v8bf*)(rowp + klocal + 16);
    return combine16(lo, hi);
}

// Fragment load from XOR-swizzled LDS tile (row stride 64, 16B chunks permuted by row&7)
static __device__ __forceinline__ v16bf frag_swz(const bf16_t* base, int r, int ks, int koff0) {
    const int c_lo = ks * 4 + (koff0 >> 3);
    const bf16_t* p0 = base + r * 64 + (((c_lo)     ^ (r & 7)) << 3);
    const bf16_t* p1 = base + r * 64 + (((c_lo + 2) ^ (r & 7)) << 3);
    return combine16(*(const v8bf*)p0, *(const v8bf*)p1);
}

static __device__ __forceinline__ v8f wmma_bf16(v16bf a, v16bf b, v8f c) {
    return __builtin_amdgcn_wmma_f32_16x16x32_bf16(false, a, false, b, (short)0, c, false, false);
}

// ---------------------------------------------------------------------------
// Conversion / layout kernels
// ---------------------------------------------------------------------------
__global__ void k_cvt(const float* __restrict__ in, bf16_t* __restrict__ out,
                      int n, float scale) {
    int i = blockIdx.x * blockDim.x + threadIdx.x;
    if (i < n) out[i] = f2bf(in[i] * scale);
}

// in: K x N row-major (fp32)  ->  out: N x K row-major (bf16, scaled)
__global__ void k_transpose_cvt(const float* __restrict__ in, bf16_t* __restrict__ out,
                                int K, int N, float scale) {
    int i = blockIdx.x * blockDim.x + threadIdx.x;
    if (i < K * N) {
        int n = i / K, k = i % K;
        out[(size_t)n * K + k] = f2bf(in[(size_t)k * N + n] * scale);
    }
}

// pos: (64 ch, 31 rel) -> posT: (32 rel, 64 ch), row 31 zero-padded
__global__ void k_pos_cvt(const float* __restrict__ in, bf16_t* __restrict__ out) {
    int t = blockIdx.x * blockDim.x + threadIdx.x;
    if (t < 32 * 64) {
        int r = t >> 6, c = t & 63;
        float v = (r < 31) ? in[c * 31 + r] : 0.0f;
        out[t] = f2bf(v);
    }
}

// ---------------------------------------------------------------------------
// 128x128 tiled bf16 WMMA GEMM:  C[M,N] = A[M,K] * BT[N,K]^T
// 128 threads = 4 waves, each wave owns a 64x64 quadrant (4x4 WMMA tiles).
// Double-buffered LDS staging via async global->LDS DMA; XOR-swizzled tiles.
// ---------------------------------------------------------------------------
template <bool OUT_F32>
__global__ __launch_bounds__(128)
void gemm128(const bf16_t* __restrict__ A, const bf16_t* __restrict__ BT,
             void* __restrict__ Cptr, int K, int ldc) {
    __shared__ bf16_t As[2][128 * 64];
    __shared__ bf16_t Bs[2][128 * 64];

    const int tid  = threadIdx.x;
    const int lane = tid & 31;
    const int wv   = tid >> 5;
    const int wm   = (wv >> 1) * 64;
    const int wn   = (wv & 1) * 64;
    const int m0   = blockIdx.y * 128;
    const int n0   = blockIdx.x * 128;
    const int koff0 = (lane & 16) ? 8 : 0;
    const int half8 = koff0;
    const int ln15  = lane & 15;

    v8f acc[4][4] = {};
    const int nch = K >> 6;   // 64-wide K chunks

    auto stage = [&](int c, int buf) {
#pragma unroll
        for (int i = 0; i < 8; ++i) {
            int v = tid + i * 128;
            int r = v >> 3, ch = v & 7;
            int ldsoff = r * 64 + ((ch ^ (r & 7)) << 3);   // xor-swizzled 16B chunks
            cp_async16(&As[buf][ldsoff], A  + (size_t)(m0 + r) * K + c * 64 + (ch << 3));
            cp_async16(&Bs[buf][ldsoff], BT + (size_t)(n0 + r) * K + c * 64 + (ch << 3));
        }
    };

    stage(0, 0);
    for (int c = 0; c < nch; ++c) {
        const int buf = c & 1;
        if (c + 1 < nch) {
            stage(c + 1, buf ^ 1);   // prefetch next chunk while this one computes
            async_wait<16>();        // retire current buffer (in-order completion)
        } else {
            async_wait<0>();
        }
        __syncthreads();
#pragma unroll
        for (int ks = 0; ks < 2; ++ks) {
            v16bf af[4], bfv[4];
#pragma unroll
            for (int t = 0; t < 4; ++t) {
                af[t]  = frag_swz(As[buf], wm + t * 16 + ln15, ks, koff0);
                bfv[t] = frag_swz(Bs[buf], wn + t * 16 + ln15, ks, koff0);
            }
#pragma unroll
            for (int mt = 0; mt < 4; ++mt)
#pragma unroll
                for (int nt = 0; nt < 4; ++nt)
                    acc[mt][nt] = wmma_bf16(af[mt], bfv[nt], acc[mt][nt]);
        }
        __syncthreads();
    }

#pragma unroll
    for (int mt = 0; mt < 4; ++mt)
#pragma unroll
        for (int nt = 0; nt < 4; ++nt)
#pragma unroll
            for (int v = 0; v < 8; ++v) {
                size_t row = (size_t)(m0 + wm + mt * 16 + v + half8);
                size_t col = (size_t)(n0 + wn + nt * 16 + ln15);
                if (OUT_F32) ((float*)Cptr)[row * ldc + col]  = acc[mt][nt][v];
                else         ((bf16_t*)Cptr)[row * ldc + col] = f2bf(acc[mt][nt][v]);
            }
}

// ---------------------------------------------------------------------------
// Halo attention core: one WG per (batch, head, block_y, block_x)
// scores = Q*K^T + relpos, softmax, out = P*V   (all matmuls on WMMA)
// ---------------------------------------------------------------------------
__global__ __launch_bounds__(128)
void halo_attn(const bf16_t* __restrict__ y,      // (32768, 1536): [q512 | kv1024]
               const bf16_t* __restrict__ poswT,  // (32, 64)
               const bf16_t* __restrict__ poshT,  // (32, 64)
               bf16_t* __restrict__ ob) {         // (32768, 512)
    __shared__ union SM {
        struct { bf16_t Q[64][72]; bf16_t Kt[256][72]; float RW[64][32]; float RH[64][32]; } a;
        struct { bf16_t P[64][256]; bf16_t V[256][64]; } b;
    } sm;

    const int ww = blockIdx.x, hh = blockIdx.y;
    const int bb = blockIdx.z >> 3, head = blockIdx.z & 7;
    const int tid = threadIdx.x, lane = tid & 31, wv = tid >> 5;
    const int koff0 = (lane & 16) ? 8 : 0;
    const int half8 = koff0;
    const int ln15  = lane & 15;

    // rel-pos B-fragments straight from global (tiny, L2-resident)
    v16bf pw[2][2], ph[2][2];
#pragma unroll
    for (int ks = 0; ks < 2; ++ks)
#pragma unroll
        for (int nt = 0; nt < 2; ++nt) {
            int row = nt * 16 + ln15;
            pw[ks][nt] = load_frag(poswT + row * 64, ks * 32 + koff0);
            ph[ks][nt] = load_frag(poshT + row * 64, ks * 32 + koff0);
        }

    // stage Q tile (64 queries x 64 ch) via async DMA
#pragma unroll
    for (int i = 0; i < 4; ++i) {
        int v = tid + i * 128;
        int qr = v >> 3, c8 = (v & 7) << 3;
        int qy = qr >> 3, qx = qr & 7;
        size_t srow = (size_t)(bb * 64 + hh * 8 + qy) * 64 + (ww * 8 + qx);
        cp_async16(&sm.a.Q[qr][c8], y + srow * 1536 + head * 64 + c8);
    }
    // stage K tile (256 halo positions x 64 ch); OOB lanes -> explicit zeros
#pragma unroll
    for (int i = 0; i < 16; ++i) {
        int v = tid + i * 128;
        int kr = v >> 3, c8 = (v & 7) << 3;
        int ky = kr >> 4, kx = kr & 15;
        int py = hh * 8 - 4 + ky, px = ww * 8 - 4 + kx;
        if (py >= 0 && py < 64 && px >= 0 && px < 64) {
            size_t srow = (size_t)(bb * 64 + py) * 64 + px;
            cp_async16(&sm.a.Kt[kr][c8], y + srow * 1536 + 512 + head * 128 + c8);
        } else {
            v8bf z = {};
            *(v8bf*)&sm.a.Kt[kr][c8] = z;
        }
    }
    async_wait<0>();
    __syncthreads();

    // scores (wave wv owns query rows wv*16..+15) + rel-pos GEMMs
    v8f sacc[16] = {};
    v8f rw[2] = {}, rh[2] = {};
#pragma unroll
    for (int ks = 0; ks < 2; ++ks) {
        const int klocal = ks * 32 + koff0;
        v16bf aq = load_frag(&sm.a.Q[wv * 16 + ln15][0], klocal);
#pragma unroll
        for (int nt = 0; nt < 16; ++nt) {
            v16bf bk = load_frag(&sm.a.Kt[nt * 16 + ln15][0], klocal);
            sacc[nt] = wmma_bf16(aq, bk, sacc[nt]);
        }
#pragma unroll
        for (int nt = 0; nt < 2; ++nt) {
            rw[nt] = wmma_bf16(aq, pw[ks][nt], rw[nt]);
            rh[nt] = wmma_bf16(aq, ph[ks][nt], rh[nt]);
        }
    }
#pragma unroll
    for (int nt = 0; nt < 2; ++nt)
#pragma unroll
        for (int v = 0; v < 8; ++v) {
            int row = wv * 16 + v + half8;
            sm.a.RW[row][nt * 16 + ln15] = rw[nt][v];
            sm.a.RH[row][nt * 16 + ln15] = rh[nt][v];
        }
    __syncthreads();

    // bias = RW[q][15+kw-qx] + RH[q][15+kh-qy]  (rel_to_abs collapsed)
#pragma unroll
    for (int nt = 0; nt < 16; ++nt)
#pragma unroll
        for (int v = 0; v < 8; ++v) {
            int q = wv * 16 + v + half8, qy = q >> 3, qx = q & 7;
            int k = nt * 16 + ln15,      kh = k >> 4, kw = k & 15;
            sacc[nt][v] += sm.a.RW[q][15 + kw - qx] + sm.a.RH[q][15 + kh - qy];
        }

    // row softmax: row lives across 16 lanes of a half x 16 acc tiles
    float invsum[8];
#pragma unroll
    for (int v = 0; v < 8; ++v) {
        float m = -3.0e38f;
#pragma unroll
        for (int nt = 0; nt < 16; ++nt) m = fmaxf(m, sacc[nt][v]);
        for (int off = 1; off < 16; off <<= 1) m = fmaxf(m, __shfl_xor(m, off, 32));
        float s = 0.0f;
#pragma unroll
        for (int nt = 0; nt < 16; ++nt) {
            float e = __expf(sacc[nt][v] - m);
            sacc[nt][v] = e;
            s += e;
        }
        for (int off = 1; off < 16; off <<= 1) s += __shfl_xor(s, off, 32);
        invsum[v] = 1.0f / s;
    }
    __syncthreads();   // everyone done with Q/Kt/RW/RH before union reuse

    // write P (bf16 probabilities) and async-stage V into the reused LDS
#pragma unroll
    for (int nt = 0; nt < 16; ++nt)
#pragma unroll
        for (int v = 0; v < 8; ++v) {
            int q = wv * 16 + v + half8;
            sm.b.P[q][nt * 16 + ln15] = f2bf(sacc[nt][v] * invsum[v]);
        }
#pragma unroll
    for (int i = 0; i < 16; ++i) {
        int v = tid + i * 128;
        int kr = v >> 3, c8 = (v & 7) << 3;
        int ky = kr >> 4, kx = kr & 15;
        int py = hh * 8 - 4 + ky, px = ww * 8 - 4 + kx;
        if (py >= 0 && py < 64 && px >= 0 && px < 64) {
            size_t srow = (size_t)(bb * 64 + py) * 64 + px;
            cp_async16(&sm.b.V[kr][c8], y + srow * 1536 + 512 + head * 128 + 64 + c8);
        } else {
            v8bf z = {};
            *(v8bf*)&sm.b.V[kr][c8] = z;
        }
    }
    async_wait<0>();
    __syncthreads();

    // out = P (64x256) * V (256x64)
    v8f oacc[4] = {};
#pragma unroll
    for (int ks = 0; ks < 8; ++ks) {
        const int klocal = ks * 32 + koff0;
        v16bf ap = load_frag(&sm.b.P[wv * 16 + ln15][0], klocal);
#pragma unroll
        for (int nt = 0; nt < 4; ++nt) {
            v16bf bv;
#pragma unroll
            for (int s = 0; s < 16; ++s)
                bv[s] = sm.b.V[klocal + (s < 8 ? s : s + 8)][nt * 16 + ln15];
            oacc[nt] = wmma_bf16(ap, bv, oacc[nt]);
        }
    }
#pragma unroll
    for (int nt = 0; nt < 4; ++nt)
#pragma unroll
        for (int v = 0; v < 8; ++v) {
            int q = wv * 16 + v + half8, qy = q >> 3, qx = q & 7;
            size_t srow = (size_t)(bb * 64 + hh * 8 + qy) * 64 + (ww * 8 + qx);
            ob[srow * 512 + head * 64 + nt * 16 + ln15] = f2bf(oacc[nt][v]);
        }
}

// ---------------------------------------------------------------------------
extern "C" void kernel_launch(void* const* d_in, const int* in_sizes, int n_in,
                              void* d_out, int out_size, void* d_ws, size_t ws_size,
                              hipStream_t stream) {
    const float* x     = (const float*)d_in[0];  // (8,64,64,512)
    const float* wq    = (const float*)d_in[1];  // (512,512)
    const float* wkv   = (const float*)d_in[2];  // (512,1024)
    const float* wout  = (const float*)d_in[3];  // (512,512)
    const float* pos_w = (const float*)d_in[4];  // (64,31)
    const float* pos_h = (const float*)d_in[5];  // (64,31)

    const size_t M = 32768;   // 8*64*64 spatial rows
    char* ws = (char*)d_ws;
    size_t off = 0;
    auto carve = [&](size_t bytes) {
        void* p = ws + off;
        off = (off + bytes + 255) & ~(size_t)255;
        return p;
    };
    bf16_t* xb     = (bf16_t*)carve(M * 512 * 2);
    bf16_t* wqkvT  = (bf16_t*)carve((size_t)1536 * 512 * 2);  // rows 0-511: wq^T*scale, 512-1535: wkv^T
    bf16_t* woutT  = (bf16_t*)carve((size_t)512 * 512 * 2);
    bf16_t* poswT  = (bf16_t*)carve((size_t)32 * 64 * 2);
    bf16_t* poshT  = (bf16_t*)carve((size_t)32 * 64 * 2);
    bf16_t* y      = (bf16_t*)carve(M * 1536 * 2);
    bf16_t* ob     = (bf16_t*)carve(M * 512 * 2);
    (void)ws_size; (void)n_in; (void)in_sizes; (void)out_size;

    const int nx = (int)(M * 512);
    k_cvt<<<(nx + 255) / 256, 256, 0, stream>>>(x, xb, nx, 1.0f);
    k_transpose_cvt<<<(512 * 512 + 255) / 256, 256, 0, stream>>>(wq, wqkvT, 512, 512, 0.125f);
    k_transpose_cvt<<<(512 * 1024 + 255) / 256, 256, 0, stream>>>(wkv, wqkvT + (size_t)512 * 512, 512, 1024, 1.0f);
    k_transpose_cvt<<<(512 * 512 + 255) / 256, 256, 0, stream>>>(wout, woutT, 512, 512, 1.0f);
    k_pos_cvt<<<8, 256, 0, stream>>>(pos_w, poswT);
    k_pos_cvt<<<8, 256, 0, stream>>>(pos_h, poshT);

    // fused QKV projection: y = xb @ [wq*scale | wkv]
    gemm128<false><<<dim3(1536 / 128, M / 128), 128, 0, stream>>>(xb, wqkvT, y, 512, 1536);

    // attention core: grid = (block_x, block_y, batch*heads)
    halo_attn<<<dim3(8, 8, 64), 128, 0, stream>>>(y, poswT, poshT, ob);

    // output projection (fp32 result)
    gemm128<true><<<dim3(512 / 128, M / 128), 128, 0, stream>>>(ob, woutT, d_out, 512, 512);
}
